// GNNPolicy_32538672234882
// MI455X (gfx1250) — matrix operand order repaired and checked by reading, not compiled, not run on hardware
//
#include <hip/hip_runtime.h>
#include <hip/hip_bf16.h>

// ---------------------------------------------------------------------------
// GNN edge-message pass for MI455X (gfx1250, wave32, WMMA).
//   PR = right @ W_left^T + b_left            (gathered by dst)
//   PL = left  @ W_right^T                    (gathered by src)
//   h  = PR[dst] + PL[src] + edge @ W_edge^T
//   msg = relu(LN(h)) @ W_final^T + b_final   <-- bf16 WMMA, fp32 accum
//   out[dst] += msg                           (f32 atomics, L2-resident)
// ---------------------------------------------------------------------------

typedef __attribute__((ext_vector_type(16))) __bf16 v16bf;
typedef __attribute__((ext_vector_type(2)))  __bf16 v2bf;
typedef __attribute__((ext_vector_type(8)))  float  v8f;

union Frag16 {
    uint4 u4[2];
    v16bf v;
};

// Packed f32x2 -> bf16x2.
// Preferred: single v_cvt_pk_bf16_f32. Fallback: round-half-up + one v_perm_b32
// (grabs the two high halves in a single VALU op). Last resort: software RNE.
#if __has_builtin(__builtin_amdgcn_cvt_pk_bf16_f32)
__device__ __forceinline__ unsigned int pack2bf(float lo, float hi) {
    v2bf r = __builtin_amdgcn_cvt_pk_bf16_f32(lo, hi);
    return __builtin_bit_cast(unsigned int, r);
}
#elif __has_builtin(__builtin_amdgcn_perm)
__device__ __forceinline__ unsigned int pack2bf(float lo, float hi) {
    unsigned int ul = __builtin_bit_cast(unsigned int, lo) + 0x8000u;
    unsigned int uh = __builtin_bit_cast(unsigned int, hi) + 0x8000u;
    // dst = { uh[31:16], ul[31:16] } : sel bytes {7,6,3,2}
    return __builtin_amdgcn_perm(uh, ul, 0x07060302u);
}
#else
__device__ __forceinline__ unsigned int f2bf_raw(float f) {
    unsigned int u = __builtin_bit_cast(unsigned int, f);
    unsigned int r = u + 0x7FFFu + ((u >> 16) & 1u);   // round-to-nearest-even
    return r >> 16;
}
__device__ __forceinline__ unsigned int pack2bf(float lo, float hi) {
    return f2bf_raw(lo) | (f2bf_raw(hi) << 16);
}
#endif

__device__ __forceinline__ v8f wmma_bf16(const Frag16& a, const Frag16& b, v8f c) {
    return __builtin_amdgcn_wmma_f32_16x16x32_bf16(
        /*neg_a=*/false, a.v, /*neg_b=*/false, b.v,
        /*c_mod=*/(short)0, c, /*reuse_a=*/false, /*reuse_b=*/false);
}

// ---------------------------------------------------------------------------
__global__ __launch_bounds__(256) void zero_f32_kernel(float* __restrict__ p, int n) {
    int i = blockIdx.x * 256 + threadIdx.x;
    if (i < n) p[i] = 0.0f;
}

// ---------------------------------------------------------------------------
// Y[m][n] = sum_k X[m][k] * W[n][k] (+ bias[n]); one 16x64 tile per wave.
__global__ __launch_bounds__(256)
void node_proj_kernel(const float* __restrict__ X, const float* __restrict__ W,
                      const float* __restrict__ bias, float* __restrict__ Y,
                      int nrows, int hasBias) {
    // W as packed bf16 pairs along K: sWU[n*32 + k/2]
    __shared__ __align__(16) unsigned int sWU[64 * 32];
    __shared__ float sB[64];

    const int tid = threadIdx.x;
#pragma unroll 2
    for (int i = tid; i < 64 * 32; i += 256) {
        int n = i >> 5, kp = i & 31;
        sWU[i] = pack2bf(W[n * 64 + kp * 2], W[n * 64 + kp * 2 + 1]);
    }
    if (tid < 64) sB[tid] = hasBias ? bias[tid] : 0.0f;
    __syncthreads();

    const int w = tid >> 5, lane = tid & 31;
    const int tile = blockIdx.x * 8 + w;
    if (tile * 16 + 16 > nrows) return;   // wave-uniform: EXEC stays all-ones

    const int row = tile * 16 + (lane & 15);
    const int hl  = lane >> 4;
    const float4* px = (const float4*)(X + (size_t)row * 64);

    // A fragments (16x32 bf16 each), CDNA5 A layout:
    // lanes 0-15 hold K {0..7,16..23}(+32kc), lanes 16-31 hold K {8..15,24..31}
    Frag16 a[2];
#pragma unroll
    for (int kc = 0; kc < 2; ++kc) {
        float4 f0 = px[kc * 8 + hl * 2 + 0];
        float4 f1 = px[kc * 8 + hl * 2 + 1];
        float4 f2 = px[kc * 8 + 4 + hl * 2 + 0];
        float4 f3 = px[kc * 8 + 4 + hl * 2 + 1];
        a[kc].u4[0] = make_uint4(pack2bf(f0.x, f0.y), pack2bf(f0.z, f0.w),
                                 pack2bf(f1.x, f1.y), pack2bf(f1.z, f1.w));
        a[kc].u4[1] = make_uint4(pack2bf(f2.x, f2.y), pack2bf(f2.z, f2.w),
                                 pack2bf(f3.x, f3.y), pack2bf(f3.z, f3.w));
    }

    const uint4* wb = (const uint4*)sWU;
    const int ncol = lane & 15;
#pragma unroll
    for (int nt = 0; nt < 4; ++nt) {
        const int n = nt * 16 + ncol;
        // B layout: lane = column n, lanes 0-15 K {0..7,16..23}(+32kc), lanes 16-31 K {8..15,24..31}
        Frag16 b0, b1;
        b0.u4[0] = wb[n * 8 + hl + 0];
        b0.u4[1] = wb[n * 8 + hl + 2];
        b1.u4[0] = wb[n * 8 + 4 + hl + 0];
        b1.u4[1] = wb[n * 8 + 4 + hl + 2];

        v8f c = {};
        c = wmma_bf16(a[0], b0, c);
        c = wmma_bf16(a[1], b1, c);

        const float bv = sB[n];
#pragma unroll
        for (int r = 0; r < 8; ++r) {
            const int m = hl * 8 + r;                      // C layout row
            Y[(size_t)(tile * 16 + m) * 64 + n] = c[r] + bv;
        }
    }
}

// ---------------------------------------------------------------------------
// Per-edge: gather + LN + ReLU + WMMA linear + atomic scatter-add.
__global__ __launch_bounds__(256)
void edge_msg_kernel(const float* __restrict__ PR, const float* __restrict__ PL,
                     const float* __restrict__ ef, const int* __restrict__ idx,
                     const float* __restrict__ We, const float* __restrict__ lg,
                     const float* __restrict__ lb, const float* __restrict__ Wf,
                     const float* __restrict__ bfin, float* __restrict__ out,
                     int E) {
    __shared__ __align__(16) unsigned int sWfU[64 * 32];   // W_final bf16 pairs
    __shared__ __align__(16) unsigned int sAu[8][16 * 32]; // per-wave A tile (bf16 pairs)
    __shared__ float sWe[128];
    __shared__ float sLg[64], sLb[64], sBf[64];
    __shared__ int   sDst[8][16];

    const int tid = threadIdx.x;
#pragma unroll 2
    for (int i = tid; i < 64 * 32; i += 256) {
        int n = i >> 5, kp = i & 31;
        sWfU[i] = pack2bf(Wf[n * 64 + kp * 2], Wf[n * 64 + kp * 2 + 1]);
    }
    if (tid < 128) sWe[tid] = We[tid];
    if (tid < 64) { sLg[tid] = lg[tid]; sLb[tid] = lb[tid]; sBf[tid] = bfin[tid]; }
    __syncthreads();

    const int w = tid >> 5, lane = tid & 31;
    const int tile = blockIdx.x * 8 + w;
    const int tileBase = tile * 16;
    if (tileBase + 16 > E) return;   // wave-uniform guard (E % 16 == 0 here)

    const int eloc = lane >> 1;      // edge within tile (2 lanes / edge)
    const int h    = lane & 1;       // feature half: h*32 .. h*32+31
    const int e    = tileBase + eloc;
    const int src  = idx[e];
    const int dst  = idx[E + e];
    if (lane < 16) sDst[w][lane] = idx[E + tileBase + lane];

    // --- gather h_i + h_j + edge projection (fp32) ---
    const float4* pi = (const float4*)(PR + (size_t)dst * 64 + h * 32);
    const float4* pj = (const float4*)(PL + (size_t)src * 64 + h * 32);
    const float e0 = ef[e * 2 + 0];
    const float e1 = ef[e * 2 + 1];

    float av[32];
#pragma unroll
    for (int q = 0; q < 8; ++q) {
        float4 a4 = pi[q];
        float4 b4 = pj[q];
        av[q * 4 + 0] = a4.x + b4.x;
        av[q * 4 + 1] = a4.y + b4.y;
        av[q * 4 + 2] = a4.z + b4.z;
        av[q * 4 + 3] = a4.w + b4.w;
    }
#pragma unroll
    for (int t = 0; t < 32; ++t) {
        const int f = h * 32 + t;
        av[t] += e0 * sWe[2 * f] + e1 * sWe[2 * f + 1];
    }

    // --- LayerNorm over 64 features (2 lanes per edge) ---
    float s = 0.0f;
#pragma unroll
    for (int t = 0; t < 32; ++t) s += av[t];
    s += __shfl_xor(s, 1, 32);
    const float mu = s * 0.015625f;

    float vs = 0.0f;
#pragma unroll
    for (int t = 0; t < 32; ++t) { const float d = av[t] - mu; vs += d * d; }
    vs += __shfl_xor(vs, 1, 32);
    const float inv = rsqrtf(vs * 0.015625f + 1e-5f);

    unsigned int up[16];
#pragma unroll
    for (int p = 0; p < 16; ++p) {
        const int f0 = h * 32 + 2 * p;
        float x0 = (av[2 * p + 0] - mu) * inv * sLg[f0]     + sLb[f0];
        float x1 = (av[2 * p + 1] - mu) * inv * sLg[f0 + 1] + sLb[f0 + 1];
        x0 = fmaxf(x0, 0.0f);
        x1 = fmaxf(x1, 0.0f);
        up[p] = pack2bf(x0, x1);
    }

    // --- stage bf16 A tile in LDS (row = edge, packed K pairs) ---
    uint4* arow = (uint4*)&sAu[w][eloc * 32 + h * 16];
    arow[0] = make_uint4(up[0],  up[1],  up[2],  up[3]);
    arow[1] = make_uint4(up[4],  up[5],  up[6],  up[7]);
    arow[2] = make_uint4(up[8],  up[9],  up[10], up[11]);
    arow[3] = make_uint4(up[12], up[13], up[14], up[15]);

    // --- read A in WMMA layout (cross-lane within the wave; DS in-order) ---
    const int row = lane & 15;
    const int hl  = lane >> 4;
    const uint4* ab = (const uint4*)&sAu[w][0];
    Frag16 a0, a1;
    a0.u4[0] = ab[row * 8 + 0 * 4 + hl + 0];
    a0.u4[1] = ab[row * 8 + 0 * 4 + hl + 2];
    a1.u4[0] = ab[row * 8 + 1 * 4 + hl + 0];
    a1.u4[1] = ab[row * 8 + 1 * 4 + hl + 2];

    // --- 16x64 @ 64x64 via 4 N-tiles x 2 K-steps, then scatter-add ---
    const uint4* wb = (const uint4*)sWfU;
#pragma unroll
    for (int nt = 0; nt < 4; ++nt) {
        const int n = nt * 16 + row;
        Frag16 b0, b1;
        b0.u4[0] = wb[n * 8 + hl + 0];
        b0.u4[1] = wb[n * 8 + hl + 2];
        b1.u4[0] = wb[n * 8 + 4 + hl + 0];
        b1.u4[1] = wb[n * 8 + 4 + hl + 2];

        v8f c = {};
        c = wmma_bf16(a0, b0, c);
        c = wmma_bf16(a1, b1, c);

        const float bv = sBf[n];
#pragma unroll
        for (int r = 0; r < 8; ++r) {
            const int m = hl * 8 + r;
            const int orow = sDst[w][m];
            atomicAdd(out + (size_t)orow * 64 + n, c[r] + bv);
        }
    }
}

// ---------------------------------------------------------------------------
extern "C" void kernel_launch(void* const* d_in, const int* in_sizes, int n_in,
                              void* d_out, int out_size, void* d_ws, size_t ws_size,
                              hipStream_t stream) {
    const float* left  = (const float*)d_in[0];
    const float* right = (const float*)d_in[1];
    const float* ef    = (const float*)d_in[2];
    const int*   idx   = (const int*)d_in[3];   // [2][E]: src then dst
    const float* Wl    = (const float*)d_in[4];
    const float* bl    = (const float*)d_in[5];
    const float* We    = (const float*)d_in[6];
    const float* Wr    = (const float*)d_in[7];
    const float* lg    = (const float*)d_in[8];
    const float* lbv   = (const float*)d_in[9];
    const float* Wf    = (const float*)d_in[10];
    const float* bfv   = (const float*)d_in[11];
    float* out = (float*)d_out;

    const int Lrows = in_sizes[0] / 64;
    const int Rrows = in_sizes[1] / 64;
    const int E     = in_sizes[2] / 2;

    float* PR = (float*)d_ws;                    // [Rrows][64]
    float* PL = PR + (size_t)Rrows * 64;         // [Lrows][64]

    // 1) zero output (harness poisons d_out)
    {
        const int n = Rrows * 64;
        zero_f32_kernel<<<(n + 255) / 256, 256, 0, stream>>>(out, n);
    }
    // 2) node projections (WMMA bf16)
    {
        const int tilesR = Rrows / 16;
        node_proj_kernel<<<(tilesR + 7) / 8, 256, 0, stream>>>(right, Wl, bl, PR, Rrows, 1);
        const int tilesL = Lrows / 16;
        node_proj_kernel<<<(tilesL + 7) / 8, 256, 0, stream>>>(left, Wr, nullptr, PL, Lrows, 0);
    }
    // 3) per-edge message + scatter-add (WMMA bf16 + f32 atomics)
    {
        const int tilesE = E / 16;
        edge_msg_kernel<<<(tilesE + 7) / 8, 256, 0, stream>>>(
            PR, PL, ef, idx, We, lg, lbv, Wf, bfv, out, E);
    }
}
